// MixtralSparseMoeBlock_47450798686724
// MI455X (gfx1250) — matrix operand order repaired and checked by reading, not compiled, not run on hardware
//
#include <hip/hip_runtime.h>

// ---------------------------------------------------------------------------
// Mixtral sparse MoE block for gfx1250 (MI455X), bf16 WMMA path.
//   x:      [T, 1024] f32   (T = 8192)
//   gate_w: [1024, 8] f32
//   w1,w3:  [8, 1024, 2048] f32   w2: [8, 2048, 1024] f32
//   out:    [T, 1024] f32
// Strategy: router -> per-expert compacted token lists; fused per-tile FFN
// (gather X, H = silu(X@W1)*(X@W3) in LDS, O = H@W2, coef-scaled scatter-add).
// Weights are pre-converted once per launch to bf16 transposed [N][K] in d_ws
// so every WMMA B-fragment is two 16B per-lane loads (fallback: direct f32).
// ---------------------------------------------------------------------------

#define EMBED 1024
#define MLP   2048
#define NEXP  8
#define TILE_M 32
#define XS_STRIDE (EMBED + 8)   // 1032 bf16 per row (pad breaks bank alignment)
#define HS_STRIDE (MLP + 8)     // 2056 bf16 per row

typedef __bf16 bf16_t;
typedef bf16_t bf16x4  __attribute__((ext_vector_type(4)));
typedef bf16_t bf16x8  __attribute__((ext_vector_type(8)));
typedef bf16_t bf16x16 __attribute__((ext_vector_type(16)));
typedef float  f32x8   __attribute__((ext_vector_type(8)));

__device__ __forceinline__ bf16_t f2bf(float f) {
  unsigned u = __builtin_bit_cast(unsigned, f);
  u += 0x7fffu + ((u >> 16) & 1u);            // round-to-nearest-even
  unsigned short h = (unsigned short)(u >> 16);
  return __builtin_bit_cast(bf16_t, h);
}

__device__ __forceinline__ f32x8 wmma_bf16(bf16x16 a, bf16x16 b, f32x8 c) {
  return __builtin_amdgcn_wmma_f32_16x16x32_bf16(
      /*neg_a=*/false, a, /*neg_b=*/false, b,
      /*c_mod=*/(short)0, c, /*reuse_a=*/false, /*reuse_b=*/false);
}

// A-matrix fragment (16x32 bf16) from LDS, contiguous K layout:
//   lane<16:  row=lane,     K = {0..15}
//   lane>=16: row=lane-16,  K = {16..31}
__device__ __forceinline__ bf16x16 load_a_frag(const bf16_t* basep, int rowBase,
                                               int k, int lane, int stride) {
  int r  = rowBase + (lane & 15);
  int hi = (lane >> 4) & 1;
  const bf16_t* p = basep + r * stride + k + hi * 16;
  bf16x8 lo = *(const bf16x8*)p;          // 16B aligned ds_load
  bf16x8 hh = *(const bf16x8*)(p + 8);
  bf16x16 a;
#pragma unroll
  for (int i = 0; i < 8; ++i) { a[i] = lo[i]; a[i + 8] = hh[i]; }
  return a;
}

// B-fragment (32x16) fast path: bf16 weights stored transposed [N][K].
// Lane l holds column col+(l&15) of the original matrix = one K-contiguous
// row of WT -> two 16B global loads, zero conversion math.
__device__ __forceinline__ bf16x16 load_b_frag_bf(const bf16_t* __restrict__ WT,
                                                  int k, int col, int ldk,
                                                  int lane) {
  int kk = k + ((lane >> 4) << 4);
  const bf16_t* p = WT + (size_t)(col + (lane & 15)) * ldk + kk;
  bf16x8 lo = *(const bf16x8*)p;
  bf16x8 hh = *(const bf16x8*)(p + 8);
  bf16x16 b;
#pragma unroll
  for (int i = 0; i < 8; ++i) { b[i] = lo[i]; b[i + 8] = hh[i]; }
  return b;
}

// B-fragment fallback: column walk of row-major f32 weights [K][ldw] with
// on-the-fly conversion (coalesced across the half-wave at each k).
__device__ __forceinline__ bf16x16 load_b_frag_f32(const float* __restrict__ W,
                                                   int k, int col, int ldw,
                                                   int lane) {
  int kk = k + ((lane >> 4) << 4);
  const float* p = W + (size_t)kk * ldw + col + (lane & 15);
  bf16x16 b;
#pragma unroll
  for (int i = 0; i < 16; ++i) b[i] = f2bf(p[(size_t)i * ldw]);
  return b;
}

// ---------------------------------------------------------------------------
__global__ void moe_zero_kernel(float* __restrict__ out, int n,
                                int* __restrict__ counts) {
  int i = blockIdx.x * blockDim.x + threadIdx.x;
  if (i < n) out[i] = 0.0f;
  if (i < NEXP) counts[i] = 0;
}

// Transpose + convert one weight tensor: src [E][K][N] f32 -> dst [E][N][K] bf16.
__global__ __launch_bounds__(256)
void weight_tc_kernel(const float* __restrict__ src, bf16_t* __restrict__ dst,
                      int K, int N) {
  __shared__ float tile[32][33];
  int e = blockIdx.z;
  const float* s = src + (size_t)e * K * N;
  bf16_t* d = dst + (size_t)e * K * N;
  int n0 = blockIdx.x * 32, k0 = blockIdx.y * 32;
  int tx = threadIdx.x & 31, ty = threadIdx.x >> 5;   // 8 rows per pass
#pragma unroll
  for (int r = 0; r < 32; r += 8)
    tile[r + ty][tx] = s[(size_t)(k0 + r + ty) * N + n0 + tx];
  __syncthreads();
#pragma unroll
  for (int r = 0; r < 32; r += 8)
    d[(size_t)(n0 + r + ty) * K + k0 + tx] = f2bf(tile[tx][r + ty]);
}

// One wave32 per token: logits -> softmax -> top-2 -> renorm -> append.
__global__ void moe_router_kernel(const float* __restrict__ x,
                                  const float* __restrict__ gate_w,
                                  int* __restrict__ counts,
                                  int* __restrict__ idxl,
                                  float* __restrict__ coefl, int T) {
  int lane = threadIdx.x & 31;
  int wave = threadIdx.x >> 5;
  int t = blockIdx.x * (blockDim.x >> 5) + wave;
  if (t >= T) return;
  const float* xr = x + (size_t)t * EMBED;
  float p[NEXP];
#pragma unroll
  for (int e = 0; e < NEXP; ++e) p[e] = 0.0f;
  for (int d = lane; d < EMBED; d += 32) {
    float xv = xr[d];
    const float* g = gate_w + (size_t)d * NEXP;
#pragma unroll
    for (int e = 0; e < NEXP; ++e) p[e] += xv * g[e];
  }
#pragma unroll
  for (int e = 0; e < NEXP; ++e) {
#pragma unroll
    for (int off = 16; off > 0; off >>= 1) p[e] += __shfl_xor(p[e], off, 32);
  }
  if (lane == 0) {
    float mx = p[0];
#pragma unroll
    for (int e = 1; e < NEXP; ++e) mx = fmaxf(mx, p[e]);
    float w[NEXP];
#pragma unroll
    for (int e = 0; e < NEXP; ++e) w[e] = __expf(p[e] - mx);
    // top-2 of softmax == top-2 of exp; renorm cancels the denominator
    int t1 = 0;
#pragma unroll
    for (int e = 1; e < NEXP; ++e) if (w[e] > w[t1]) t1 = e;
    int t2 = (t1 == 0) ? 1 : 0;
#pragma unroll
    for (int e = 0; e < NEXP; ++e) if (e != t1 && w[e] > w[t2]) t2 = e;
    float s2 = w[t1] + w[t2];
    float c1 = w[t1] / s2, c2 = w[t2] / s2;
    int p1 = atomicAdd(&counts[t1], 1);
    idxl[t1 * T + p1] = t; coefl[t1 * T + p1] = c1;
    int p2 = atomicAdd(&counts[t2], 1);
    idxl[t2 * T + p2] = t; coefl[t2 * T + p2] = c2;
  }
}

// Fused expert FFN over one 32-token tile of one expert's routed list.
template <bool BW>
__global__ __launch_bounds__(256)
void moe_ffn_kernel(const float* __restrict__ x,
                    const float* __restrict__ w1f,
                    const float* __restrict__ w2f,
                    const float* __restrict__ w3f,
                    const bf16_t* __restrict__ w1T,
                    const bf16_t* __restrict__ w3T,
                    const bf16_t* __restrict__ w2T,
                    const int* __restrict__ counts,
                    const int* __restrict__ idxl,
                    const float* __restrict__ coefl,
                    float* __restrict__ out, int T) {
  extern __shared__ char smem[];
  bf16_t* xs    = (bf16_t*)smem;                                   // [32][1032]
  bf16_t* hs    = (bf16_t*)(smem + TILE_M * XS_STRIDE * 2);        // [32][2056]
  int*    idx_s = (int*)(smem + TILE_M * XS_STRIDE * 2 + TILE_M * HS_STRIDE * 2);
  float*  cf_s  = (float*)(idx_s + TILE_M);

  int e = blockIdx.y;
  int cnt = counts[e];
  int base = blockIdx.x * TILE_M;
  if (base >= cnt) return;                     // block-uniform early exit

  int tid = threadIdx.x, lane = tid & 31, wave = tid >> 5;

  if (tid < TILE_M) {
    int g = base + tid;
    if (g < cnt) { idx_s[tid] = idxl[e * T + g]; cf_s[tid] = coefl[e * T + g]; }
    else         { idx_s[tid] = idxl[e * T + base]; cf_s[tid] = 0.0f; }
  }
  __syncthreads();

  // Gather X rows -> LDS bf16, float4 loads + packed bf16x4 stores
  for (int j = tid; j < TILE_M * (EMBED / 4); j += 256) {
    int r = j >> 8;                 // EMBED/4 == 256
    int d = (j & 255) << 2;
    const float4 v = *(const float4*)(x + (size_t)idx_s[r] * EMBED + d);
    bf16x4 o;
    o[0] = f2bf(v.x); o[1] = f2bf(v.y); o[2] = f2bf(v.z); o[3] = f2bf(v.w);
    *(bf16x4*)(xs + r * XS_STRIDE + d) = o;
  }
  __syncthreads();

  const float*  w1fe = w1f + (size_t)e * EMBED * MLP;
  const float*  w3fe = w3f + (size_t)e * EMBED * MLP;
  const float*  w2fe = w2f + (size_t)e * MLP * EMBED;
  const bf16_t* w1Te = BW ? w1T + (size_t)e * EMBED * MLP : nullptr;
  const bf16_t* w3Te = BW ? w3T + (size_t)e * EMBED * MLP : nullptr;
  const bf16_t* w2Te = BW ? w2T + (size_t)e * MLP * EMBED : nullptr;

  int nsub = lane & 15;
  int hsel = lane >> 4;   // C layout: VGPR i holds row i + 8*hsel

  // ---- Phase 1: H = silu(X@W1) * (X@W3), 256 cols per wave ----
  for (int cc = 0; cc < 16; ++cc) {
    int col = wave * 256 + cc * 16;
    if (BW) {   // warm L2/L0 for the two weight panels of this chunk
      __builtin_prefetch(w1Te + (size_t)(col + nsub) * EMBED, 0, 1);
      __builtin_prefetch(w3Te + (size_t)(col + nsub) * EMBED, 0, 1);
    }
    f32x8 a1a = {}, a1b = {}, a3a = {}, a3b = {};
    for (int k = 0; k < EMBED; k += 32) {
      bf16x16 A0 = load_a_frag(xs, 0, k, lane, XS_STRIDE);
      bf16x16 A1 = load_a_frag(xs, 16, k, lane, XS_STRIDE);
      bf16x16 B1 = BW ? load_b_frag_bf(w1Te, k, col, EMBED, lane)
                      : load_b_frag_f32(w1fe, k, col, MLP, lane);
      bf16x16 B3 = BW ? load_b_frag_bf(w3Te, k, col, EMBED, lane)
                      : load_b_frag_f32(w3fe, k, col, MLP, lane);
      a1a = wmma_bf16(A0, B1, a1a);
      a1b = wmma_bf16(A1, B1, a1b);
      a3a = wmma_bf16(A0, B3, a3a);
      a3b = wmma_bf16(A1, B3, a3b);
    }
#pragma unroll
    for (int i = 0; i < 8; ++i) {
      int m0 = i + 8 * hsel;
      float g0 = a1a[i];
      float h0 = g0 * __builtin_amdgcn_rcpf(1.0f + __expf(-g0)) * a3a[i];
      hs[m0 * HS_STRIDE + col + nsub] = f2bf(h0);
      float g1 = a1b[i];
      float h1 = g1 * __builtin_amdgcn_rcpf(1.0f + __expf(-g1)) * a3b[i];
      hs[(m0 + 16) * HS_STRIDE + col + nsub] = f2bf(h1);
    }
  }
  __syncthreads();

  // ---- Phase 2: O = H @ W2, scale by coef, scatter-add ----
  for (int cc = 0; cc < 8; ++cc) {
    int col = wave * 128 + cc * 16;
    if (BW) __builtin_prefetch(w2Te + (size_t)(col + nsub) * MLP, 0, 1);
    f32x8 oa = {}, ob = {};
    for (int k = 0; k < MLP; k += 32) {
      bf16x16 A0 = load_a_frag(hs, 0, k, lane, HS_STRIDE);
      bf16x16 A1 = load_a_frag(hs, 16, k, lane, HS_STRIDE);
      bf16x16 B2 = BW ? load_b_frag_bf(w2Te, k, col, MLP, lane)
                      : load_b_frag_f32(w2fe, k, col, EMBED, lane);
      oa = wmma_bf16(A0, B2, oa);
      ob = wmma_bf16(A1, B2, ob);
    }
#pragma unroll
    for (int i = 0; i < 8; ++i) {
      int m0 = i + 8 * hsel;
      float v0 = oa[i] * cf_s[m0];
      __hip_atomic_fetch_add(out + (size_t)idx_s[m0] * EMBED + col + nsub, v0,
                             __ATOMIC_RELAXED, __HIP_MEMORY_SCOPE_AGENT);
      int m1 = m0 + 16;
      float v1 = ob[i] * cf_s[m1];
      __hip_atomic_fetch_add(out + (size_t)idx_s[m1] * EMBED + col + nsub, v1,
                             __ATOMIC_RELAXED, __HIP_MEMORY_SCOPE_AGENT);
    }
  }
}

// ---------------------------------------------------------------------------
extern "C" void kernel_launch(void* const* d_in, const int* in_sizes, int n_in,
                              void* d_out, int out_size, void* d_ws, size_t ws_size,
                              hipStream_t stream) {
  const float* x      = (const float*)d_in[0];
  const float* gate_w = (const float*)d_in[1];
  const float* w1     = (const float*)d_in[2];
  const float* w2     = (const float*)d_in[3];
  const float* w3     = (const float*)d_in[4];
  float* out = (float*)d_out;

  const int T = in_sizes[0] / EMBED;          // 8192 tokens

  // Workspace layout:
  //  [counts: 8 int | pad to 256B | idx: NEXP*T int | coef: NEXP*T f32 |
  //   pad to 256B | w1T bf16 | w3T bf16 | w2T bf16]
  int*   counts = (int*)d_ws;
  int*   idxl   = (int*)((char*)d_ws + 256);
  float* coefl  = (float*)((char*)d_ws + 256 + (size_t)NEXP * T * sizeof(int));
  size_t off_w  = (256 + (size_t)NEXP * T * 8 + 255) & ~(size_t)255;
  size_t wbytes = (size_t)NEXP * EMBED * MLP * sizeof(bf16_t);   // 33.5 MB each
  bf16_t* w1T = (bf16_t*)((char*)d_ws + off_w);
  bf16_t* w3T = (bf16_t*)((char*)d_ws + off_w + wbytes);
  bf16_t* w2T = (bf16_t*)((char*)d_ws + off_w + 2 * wbytes);
  const bool bw = ws_size >= off_w + 3 * wbytes;

  // 1) zero out + counters (out is accumulated with atomics)
  moe_zero_kernel<<<(out_size + 255) / 256, 256, 0, stream>>>(out, out_size,
                                                              counts);

  // 2) router: one wave per token
  moe_router_kernel<<<(T + 7) / 8, 256, 0, stream>>>(x, gate_w, counts, idxl,
                                                     coefl, T);

  // 3) pre-convert weights to transposed bf16 (one-shot ~100MB HBM traffic)
  if (bw) {
    weight_tc_kernel<<<dim3(MLP / 32, EMBED / 32, NEXP), 256, 0, stream>>>(
        w1, w1T, EMBED, MLP);
    weight_tc_kernel<<<dim3(MLP / 32, EMBED / 32, NEXP), 256, 0, stream>>>(
        w3, w3T, EMBED, MLP);
    weight_tc_kernel<<<dim3(EMBED / 32, MLP / 32, NEXP), 256, 0, stream>>>(
        w2, w2T, MLP, EMBED);
  }

  // 4) expert FFN: worst-case grid, blocks past count[e] exit immediately
  size_t smem = (size_t)TILE_M * XS_STRIDE * 2 + (size_t)TILE_M * HS_STRIDE * 2 +
                (size_t)TILE_M * sizeof(int) + (size_t)TILE_M * sizeof(float);
  dim3 grid(T / TILE_M, NEXP);
  if (bw) {
    moe_ffn_kernel<true><<<grid, 256, smem, stream>>>(
        x, w1, w2, w3, w1T, w3T, w2T, counts, idxl, coefl, out, T);
  } else {
    moe_ffn_kernel<false><<<grid, 256, smem, stream>>>(
        x, w1, w2, w3, nullptr, nullptr, nullptr, counts, idxl, coefl, out, T);
  }
}
